// PointTransformerLayer_13975823581708
// MI455X (gfx1250) — compile-verified, matrix operand order checked by reading, not run on hardware
//
#include <hip/hip_runtime.h>
#include <hip/hip_bf16.h>
#include <stdint.h>

typedef __attribute__((ext_vector_type(16))) _Float16 v16h;
typedef __attribute__((ext_vector_type(8)))  _Float16 v8h;
typedef __attribute__((ext_vector_type(8)))  float    v8f;

#define NPTS  32768
#define NSAMP 16
#define MROWS (NPTS * NSAMP)
#define PI_F  3.14159265358979323846f

// ---------------------------------------------------------------------------
// WMMA helpers (CDNA5 ISA 7.12.2 layouts):
//   A (16x32 f16): lane = m + 16*hi; half j -> K = (j&7) + 16*(j>>3) + 8*hi
//     => per lane, two contiguous 8-half runs at k0+8*hi and k0+16+8*hi
//   B (32x16 f16): symmetric; we PRE-SWIZZLE B into fragment order in LDS so
//     each lane reads one contiguous 32-byte chunk per K-step (ds_load_b128).
//   C/D (16x16 f32): VGPR r -> row = r + 8*hi, col = lane&15
// ---------------------------------------------------------------------------

// Scatter W (fp32, KTOT x 64 row-major) into fragment-ordered f16 LDS.
// Fragment address: ((ctile*KS + ks)*32 + lane)*16 + j
template<int KTOT>
__device__ __forceinline__ void stage_b_frag(const float* __restrict__ W,
                                             _Float16* __restrict__ BsF, int tid)
{
  constexpr int KS = KTOT / 32;
  for (int i = tid; i < KTOT * 64; i += 256) {
    const int kIdx = i >> 6, nIdx = i & 63;
    const int ks = kIdx >> 5, kr = kIdx & 31;
    const int j    = (kr & 7) | ((kr >> 4) << 3);
    const int hi   = (kr >> 3) & 1;
    const int lane = (nIdx & 15) | (hi << 4);
    const int ct   = nIdx >> 4;
    BsF[(((ct * KS + ks) * 32 + lane) << 4) | j] = (_Float16)W[i];
  }
}

template<int KTOT>
__device__ __forceinline__ v8f wmma_tile_frag(const _Float16* As, int ldA,
                                              const _Float16* BsF,
                                              int mBase, int ctile, int lane)
{
  constexpr int KS = KTOT / 32;
  v8f acc = {0.f, 0.f, 0.f, 0.f, 0.f, 0.f, 0.f, 0.f};
  const int m  = lane & 15;
  const int hi = lane >> 4;
  const _Float16* arow = As + (mBase + m) * ldA + 8 * hi;
#pragma unroll
  for (int ks = 0; ks < KS; ++ks) {
    const v8h alo = *(const v8h*)(arow + ks * 32);
    const v8h ahi = *(const v8h*)(arow + ks * 32 + 16);
    v16h a;
#pragma unroll
    for (int j = 0; j < 8; ++j) { a[j] = alo[j]; a[j + 8] = ahi[j]; }
    const v16h b = *(const v16h*)(BsF + (((ctile * KS + ks) * 32 + lane) << 4));
    acc = __builtin_amdgcn_wmma_f32_16x16x32_f16(false, a, false, b,
                                                 (short)0, acc, false, false);
  }
  return acc;
}

__device__ __forceinline__ float wave_sum32(float v) {
#pragma unroll
  for (int mm = 16; mm >= 1; mm >>= 1) v += __shfl_xor(v, mm, 32);
  return v;
}
__device__ __forceinline__ float oct_sum(float v) {
#pragma unroll
  for (int mm = 4; mm >= 1; mm >>= 1) v += __shfl_xor(v, mm, 32);
  return v;
}

// ---------------------------------------------------------------------------
// K0: wpq row-means for q2, zero BN accumulators
// ---------------------------------------------------------------------------
__global__ void k0_prep(const float* __restrict__ Wpq, const float* __restrict__ bpq,
                        float* __restrict__ wpqbar, float* __restrict__ q2bias,
                        float* __restrict__ bnAcc /* 4*64 floats */)
{
  int t = threadIdx.x;
  if (t < 64) {
    float s = 0.f;
    for (int c = 0; c < 64; ++c) s += Wpq[t * 64 + c];
    wpqbar[t] = s * (1.f / 64.f);
  }
  if (t == 64) {
    float s = 0.f;
    for (int c = 0; c < 64; ++c) s += bpq[c];
    q2bias[0] = s * (1.f / 64.f);
  }
  for (int i = t; i < 256; i += blockDim.x) bnAcc[i] = 0.f;
}

// ---------------------------------------------------------------------------
// K1: q,k,v,x2 = x @ {Wq,Wk,Wv,Wx} + bias   (WMMA, fp32 out)
// grid (NPTS/32, 4), block 256
// ---------------------------------------------------------------------------
__global__ void k1_qkvx(const float* __restrict__ x,
                        const float* __restrict__ Wq, const float* __restrict__ bq, float* __restrict__ oq,
                        const float* __restrict__ Wk, const float* __restrict__ bk, float* __restrict__ ok,
                        const float* __restrict__ Wv, const float* __restrict__ bv, float* __restrict__ ov,
                        const float* __restrict__ Wx, const float* __restrict__ bx, float* __restrict__ ox)
{
  __shared__ __attribute__((aligned(16))) _Float16 As[32 * 72];
  __shared__ __attribute__((aligned(32))) _Float16 BsF[64 * 64];
  __shared__ float sBias[64];

  const float* W; const float* bias; float* out;
  switch (blockIdx.y) {
    case 0:  W = Wq; bias = bq; out = oq; break;
    case 1:  W = Wk; bias = bk; out = ok; break;
    case 2:  W = Wv; bias = bv; out = ov; break;
    default: W = Wx; bias = bx; out = ox; break;
  }
  const int tid = threadIdx.x;
  const size_t r0 = (size_t)blockIdx.x * 32;

  for (int i = tid; i < 32 * 64; i += 256)
    As[(i >> 6) * 72 + (i & 63)] = (_Float16)x[r0 * 64 + i];
  stage_b_frag<64>(W, BsF, tid);
  if (tid < 64) sBias[tid] = bias[tid];
  __syncthreads();

  const int wv = tid >> 5, lane = tid & 31;
  const int mBase = (wv & 1) * 16, ctile = wv >> 1;
  v8f acc = wmma_tile_frag<64>(As, 72, BsF, mBase, ctile, lane);

  const int hi = lane >> 4, nn = lane & 15;
#pragma unroll
  for (int r = 0; r < 8; ++r) {
    const int row = mBase + r + 8 * hi;
    const int col = ctile * 16 + nn;
    out[(r0 + row) * 64 + col] = acc[r] + sBias[col];
  }
}

// ---------------------------------------------------------------------------
// K2: pr[n,t,:] = relu(LN(sph6 @ pW1 + pb1)) @ pW2 + pb2  -> f16 buffer
// thread per (n,t); grid MROWS/256, block 256
// ---------------------------------------------------------------------------
__global__ void k2_pr(const float* __restrict__ p, const int* __restrict__ idx,
                      const float* __restrict__ W1, const float* __restrict__ b1,
                      const float* __restrict__ lng, const float* __restrict__ lnb,
                      const float* __restrict__ W2, const float* __restrict__ b2,
                      _Float16* __restrict__ prbuf)
{
  __shared__ float sW1[36], sB1[6], sG[6], sB[6], sW2[6 * 64], sB2[64];
  const int tid = threadIdx.x;
  if (tid < 36) sW1[tid] = W1[tid];
  if (tid < 6) { sB1[tid] = b1[tid]; sG[tid] = lng[tid]; sB[tid] = lnb[tid]; }
  for (int i = tid; i < 384; i += 256) sW2[i] = W2[i];
  if (tid < 64) sB2[tid] = b2[tid];
  __syncthreads();

  const size_t e = (size_t)blockIdx.x * 256 + tid;
  const int n  = (int)(e >> 4);
  const int gv = idx[e];

  const float px = p[n * 3], py = p[n * 3 + 1], pz = p[n * 3 + 2];
  const float rx = p[gv * 3] - px, ry = p[gv * 3 + 1] - py, rz = p[gv * 3 + 2] - pz;
  const float rho = sqrtf(rx * rx + ry * ry + rz * rz);
  float theta = 0.f;
  if (rho > 0.f) theta = acosf(fminf(fmaxf(rz / rho, -1.f), 1.f)) * (1.f / PI_F);
  const float phi = atan2f(ry, rx) * (1.f / (2.f * PI_F)) + 0.5f;
  const float in6[6] = {rx, ry, rz, rho, theta, phi};

  float h[6];
#pragma unroll
  for (int o = 0; o < 6; ++o) {
    float a = sB1[o];
#pragma unroll
    for (int j = 0; j < 6; ++j) a += in6[j] * sW1[j * 6 + o];
    h[o] = a;
  }
  float s = 0.f, sq = 0.f;
#pragma unroll
  for (int o = 0; o < 6; ++o) { s += h[o]; sq += h[o] * h[o]; }
  const float mean = s * (1.f / 6.f);
  const float var  = sq * (1.f / 6.f) - mean * mean;
  const float inv  = rsqrtf(var + 1e-5f);
#pragma unroll
  for (int o = 0; o < 6; ++o) h[o] = fmaxf((h[o] - mean) * inv * sG[o] + sB[o], 0.f);

  for (int c = 0; c < 64; ++c) {
    float a = sB2[c];
#pragma unroll
    for (int j = 0; j < 6; ++j) a += h[j] * sW2[j * 64 + c];
    prbuf[e * 64 + c] = (_Float16)a;
  }
}

// ---------------------------------------------------------------------------
// K3: fw chain + softmax + feat einsum + BN(feat) stats; wave per point
// grid NPTS/8, block 256
// ---------------------------------------------------------------------------
__global__ void k3_feat(const float* __restrict__ qb, const float* __restrict__ kb,
                        const float* __restrict__ vb, const _Float16* __restrict__ prbuf,
                        const int* __restrict__ idx,
                        const float* __restrict__ ln1g, const float* __restrict__ ln1b,
                        const float* __restrict__ W1,  const float* __restrict__ b1,
                        const float* __restrict__ ln2g, const float* __restrict__ ln2b,
                        const float* __restrict__ W2,  const float* __restrict__ b2,
                        float* __restrict__ featbuf,
                        float* __restrict__ bnsum, float* __restrict__ bnsq)
{
  __shared__ float sQ[8][64];
  __shared__ float sVP[8][NSAMP][64];
  __shared__ float sH[8][64];
  __shared__ float sLog[8][NSAMP][8];
  __shared__ float sW1[512], sG1[64], sBn1[64], sW2s[64];
  __shared__ float sB1[8], sG2[8], sBn2[8], sB2w[8];
  __shared__ float sBNs[64], sBNq[64];

  const int tid = threadIdx.x, wv = tid >> 5, ln = tid & 31;
  const int n = blockIdx.x * 8 + wv;

  for (int i = tid; i < 512; i += 256) sW1[i] = W1[i];
  if (tid < 64) { sG1[tid] = ln1g[tid]; sBn1[tid] = ln1b[tid]; sW2s[tid] = W2[tid]; }
  if (tid < 8) { sB1[tid] = b1[tid]; sG2[tid] = ln2g[tid]; sBn2[tid] = ln2b[tid]; sB2w[tid] = b2[tid]; }
  if (tid < 64) { sBNs[tid] = 0.f; sBNq[tid] = 0.f; }
  sQ[wv][ln]      = qb[(size_t)n * 64 + ln];
  sQ[wv][ln + 32] = qb[(size_t)n * 64 + ln + 32];
  __syncthreads();

  for (int t = 0; t < NSAMP; ++t) {
    const int gv = idx[n * NSAMP + t];
    if (t + 1 < NSAMP) {
      const int gvn = idx[n * NSAMP + t + 1];
      __builtin_prefetch(kb + (size_t)gvn * 64, 0, 0);
      __builtin_prefetch(vb + (size_t)gvn * 64, 0, 0);
    }
    const size_t row = (size_t)(n * NSAMP + t) * 64;
    const int c0 = ln, c1 = ln + 32;
    const float pr0 = (float)prbuf[row + c0];
    const float pr1 = (float)prbuf[row + c1];
    const float h0 = kb[(size_t)gv * 64 + c0] - sQ[wv][c0] + pr0;
    const float h1 = kb[(size_t)gv * 64 + c1] - sQ[wv][c1] + pr1;
    sVP[wv][t][c0] = vb[(size_t)gv * 64 + c0] + pr0;
    sVP[wv][t][c1] = vb[(size_t)gv * 64 + c1] + pr1;

    // LN over 64
    const float s  = wave_sum32(h0 + h1);
    const float sq = wave_sum32(h0 * h0 + h1 * h1);
    const float mean = s * (1.f / 64.f);
    const float var  = sq * (1.f / 64.f) - mean * mean;
    const float inv  = rsqrtf(var + 1e-5f);
    sH[wv][c0] = fmaxf((h0 - mean) * inv * sG1[c0] + sBn1[c0], 0.f);
    sH[wv][c1] = fmaxf((h1 - mean) * inv * sG1[c1] + sBn1[c1], 0.f);
    __syncthreads();

    // 64->8, LN over 8, relu, 8->8
    float lg = 0.f;
    if (ln < 8) {
      for (int c = 0; c < 64; ++c) lg += sH[wv][c] * sW1[c * 8 + ln];
      lg += sB1[ln];
    }
    const float s2  = oct_sum((ln < 8) ? lg : 0.f);
    const float q2s = oct_sum((ln < 8) ? lg * lg : 0.f);
    const float m2   = s2 * 0.125f;
    const float v2v  = q2s * 0.125f - m2 * m2;
    const float inv2 = rsqrtf(v2v + 1e-5f);
    float hn2 = 0.f;
    if (ln < 8) hn2 = fmaxf((lg - m2) * inv2 * sG2[ln] + sBn2[ln], 0.f);
    float outj = 0.f;
#pragma unroll
    for (int i = 0; i < 8; ++i) {
      const float hv = __shfl(hn2, i, 8);
      if (ln < 8) outj += hv * sW2s[i * 8 + ln];
    }
    if (ln < 8) sLog[wv][t][ln] = outj + sB2w[ln];
    __syncthreads();
  }

  // softmax over NSAMP per i
  if (ln < 8) {
    float mx = -1e30f;
    for (int t = 0; t < NSAMP; ++t) mx = fmaxf(mx, sLog[wv][t][ln]);
    float ssum = 0.f;
    for (int t = 0; t < NSAMP; ++t) { const float e = __expf(sLog[wv][t][ln] - mx); sLog[wv][t][ln] = e; ssum += e; }
    const float invs = 1.f / ssum;
    for (int t = 0; t < NSAMP; ++t) sLog[wv][t][ln] *= invs;
  }
  __syncthreads();

#pragma unroll
  for (int cc = 0; cc < 2; ++cc) {
    const int c = ln + 32 * cc;
    float acc = 0.f;
    for (int t = 0; t < NSAMP; ++t) acc += sVP[wv][t][c] * sLog[wv][t][c & 7];
    featbuf[(size_t)n * 64 + c] = acc;
    atomicAdd(&sBNs[c], acc);
    atomicAdd(&sBNq[c], acc * acc);
  }
  __syncthreads();
  if (tid < 64) { atomicAdd(&bnsum[tid], sBNs[tid]); atomicAdd(&bnsq[tid], sBNq[tid]); }
}

// ---------------------------------------------------------------------------
// K4a: B1 = relu(LN(B1)) in place; wave per row; grid MROWS/8, block 256
// ---------------------------------------------------------------------------
__global__ void k4a_lnrelu(_Float16* __restrict__ B1,
                           const float* __restrict__ g, const float* __restrict__ b)
{
  __shared__ float sG[64], sB[64];
  const int tid = threadIdx.x, wv = tid >> 5, ln = tid & 31;
  if (tid < 64) { sG[tid] = g[tid]; sB[tid] = b[tid]; }
  __syncthreads();
  const size_t row = ((size_t)blockIdx.x * 8 + wv) * 64;
  const float h0 = (float)B1[row + ln];
  const float h1 = (float)B1[row + ln + 32];
  const float s  = wave_sum32(h0 + h1);
  const float sq = wave_sum32(h0 * h0 + h1 * h1);
  const float mean = s * (1.f / 64.f);
  const float var  = sq * (1.f / 64.f) - mean * mean;
  const float inv  = rsqrtf(var + 1e-5f);
  B1[row + ln]      = (_Float16)fmaxf((h0 - mean) * inv * sG[ln] + sB[ln], 0.f);
  B1[row + ln + 32] = (_Float16)fmaxf((h1 - mean) * inv * sG[ln + 32] + sB[ln + 32], 0.f);
}

// ---------------------------------------------------------------------------
// K4b: B2 = relu(LN2(B1 @ p2_W + p2_b))  (WMMA + fused LN epilogue)
// A tile staged via CDNA5 async global->LDS copy. grid MROWS/32, block 256
// ---------------------------------------------------------------------------
__global__ void k4b_gemm_ln(const _Float16* __restrict__ A, const float* __restrict__ W,
                            const float* __restrict__ bias, const float* __restrict__ g,
                            const float* __restrict__ bln, _Float16* __restrict__ Out)
{
  __shared__ __attribute__((aligned(16))) _Float16 As[32 * 64];
  __shared__ __attribute__((aligned(32))) _Float16 BsF[64 * 64];
  __shared__ float Cs[32 * 72];
  __shared__ float sBias[64], sG[64], sB[64];
  const int tid = threadIdx.x;
  const size_t r0 = (size_t)blockIdx.x * 32;

  // async copy: 256 lanes x 16B = full 32x64 f16 tile (ASYNCcnt tracked)
  {
    const unsigned int       loff = (unsigned int)(uintptr_t)(&As[tid * 8]);
    const unsigned long long ga   = (unsigned long long)(uintptr_t)(A + r0 * 64 + tid * 8);
    asm volatile("global_load_async_to_lds_b128 %0, %1, off" :: "v"(loff), "v"(ga) : "memory");
  }
  stage_b_frag<64>(W, BsF, tid);
  if (tid < 64) { sBias[tid] = bias[tid]; sG[tid] = g[tid]; sB[tid] = bln[tid]; }
  asm volatile("s_wait_asynccnt 0" ::: "memory");
  __syncthreads();

  const int wv = tid >> 5, lane = tid & 31;
  const int mBase = (wv & 1) * 16, ctile = wv >> 1;
  v8f acc = wmma_tile_frag<64>(As, 64, BsF, mBase, ctile, lane);
  const int hi = lane >> 4, nn = lane & 15;
#pragma unroll
  for (int r = 0; r < 8; ++r)
    Cs[(mBase + r + 8 * hi) * 72 + ctile * 16 + nn] = acc[r] + sBias[ctile * 16 + nn];
  __syncthreads();

  if (tid < 32) {
    const int row = tid;
    float s = 0.f, sq = 0.f;
    for (int c = 0; c < 64; ++c) { const float h = Cs[row * 72 + c]; s += h; sq += h * h; }
    const float mean = s * (1.f / 64.f);
    const float var  = sq * (1.f / 64.f) - mean * mean;
    const float inv  = rsqrtf(var + 1e-5f);
    for (int c = 0; c < 64; ++c) {
      const float h = (Cs[row * 72 + c] - mean) * inv * sG[c] + sB[c];
      Out[(r0 + row) * 64 + c] = (_Float16)fmaxf(h, 0.f);
    }
  }
}

// ---------------------------------------------------------------------------
// K4d: k2 = B2 @ Wpk + bpk (WMMA, f16 out -> B1); q2 = B2 . wpqbar + q2bias
// A tile staged via CDNA5 async global->LDS copy. grid MROWS/32, block 256
// ---------------------------------------------------------------------------
__global__ void k4d_gemm_q2(const _Float16* __restrict__ A, const float* __restrict__ Wpk,
                            const float* __restrict__ bpk, _Float16* __restrict__ K2out,
                            const float* __restrict__ wpqbar, const float* __restrict__ q2bias,
                            float* __restrict__ q2out)
{
  __shared__ __attribute__((aligned(16))) _Float16 As[32 * 64];
  __shared__ __attribute__((aligned(32))) _Float16 BsF[64 * 64];
  __shared__ float sBias[64], sQW[64];
  const int tid = threadIdx.x;
  const size_t r0 = (size_t)blockIdx.x * 32;

  {
    const unsigned int       loff = (unsigned int)(uintptr_t)(&As[tid * 8]);
    const unsigned long long ga   = (unsigned long long)(uintptr_t)(A + r0 * 64 + tid * 8);
    asm volatile("global_load_async_to_lds_b128 %0, %1, off" :: "v"(loff), "v"(ga) : "memory");
  }
  stage_b_frag<64>(Wpk, BsF, tid);
  if (tid < 64) { sBias[tid] = bpk[tid]; sQW[tid] = wpqbar[tid]; }
  asm volatile("s_wait_asynccnt 0" ::: "memory");
  __syncthreads();

  const int wv = tid >> 5, lane = tid & 31;
  const int mBase = (wv & 1) * 16, ctile = wv >> 1;
  v8f acc = wmma_tile_frag<64>(As, 64, BsF, mBase, ctile, lane);
  const int hi = lane >> 4, nn = lane & 15;
#pragma unroll
  for (int r = 0; r < 8; ++r) {
    const int row = mBase + r + 8 * hi;
    K2out[(r0 + row) * 64 + ctile * 16 + nn] = (_Float16)(acc[r] + sBias[ctile * 16 + nn]);
  }
  if (tid < 32) {
    float a = q2bias[0];
    for (int j = 0; j < 64; ++j) a += (float)As[tid * 64 + j] * sQW[j];
    q2out[r0 + tid] = a;
  }
}

// ---------------------------------------------------------------------------
// K5: pw chain + softmax + post einsum (v2 matvec fused) + BN(post) stats
// grid NPTS/8, block 256
// ---------------------------------------------------------------------------
__global__ void k5_post(const float* __restrict__ x2b, const _Float16* __restrict__ k2buf,
                        const _Float16* __restrict__ t2buf, const float* __restrict__ q2buf,
                        const int* __restrict__ idx,
                        const float* __restrict__ Wpv, const float* __restrict__ bpv,
                        const float* __restrict__ ln1g, const float* __restrict__ ln1b,
                        const float* __restrict__ W1,  const float* __restrict__ b1,
                        const float* __restrict__ ln2g, const float* __restrict__ ln2b,
                        const float* __restrict__ W2,  const float* __restrict__ b2,
                        float* __restrict__ postbuf,
                        float* __restrict__ bnsum, float* __restrict__ bnsq)
{
  __shared__ float sVP[8][NSAMP][64];
  __shared__ float sT[8][64];
  __shared__ float sH[8][64];
  __shared__ float sLog[8][NSAMP][8];
  __shared__ float sWPV[64 * 64];
  __shared__ float sBPV[64];
  __shared__ float sW1[512], sG1[64], sBn1[64], sW2s[64];
  __shared__ float sB1[8], sG2[8], sBn2[8], sB2w[8];
  __shared__ float sBNs[64], sBNq[64];

  const int tid = threadIdx.x, wv = tid >> 5, ln = tid & 31;
  const int n = blockIdx.x * 8 + wv;

  for (int i = tid; i < 64 * 64; i += 256) sWPV[i] = Wpv[i];
  for (int i = tid; i < 512; i += 256) sW1[i] = W1[i];
  if (tid < 64) { sBPV[tid] = bpv[tid]; sG1[tid] = ln1g[tid]; sBn1[tid] = ln1b[tid]; sW2s[tid] = W2[tid]; }
  if (tid < 8) { sB1[tid] = b1[tid]; sG2[tid] = ln2g[tid]; sBn2[tid] = ln2b[tid]; sB2w[tid] = b2[tid]; }
  if (tid < 64) { sBNs[tid] = 0.f; sBNq[tid] = 0.f; }
  __syncthreads();

  for (int t = 0; t < NSAMP; ++t) {
    const int gv = idx[n * NSAMP + t];
    if (t + 1 < NSAMP) {
      const int gvn = idx[n * NSAMP + t + 1];
      __builtin_prefetch(x2b + (size_t)gvn * 64, 0, 0);
    }
    const size_t row = (size_t)(n * NSAMP + t) * 64;
    sT[wv][ln]      = (float)t2buf[row + ln];
    sT[wv][ln + 32] = (float)t2buf[row + ln + 32];
    __syncthreads();

    const float q2v = q2buf[n * NSAMP + t];
    const int c0 = ln, c1 = ln + 32;
    const float xg0 = x2b[(size_t)gv * 64 + c0];
    const float xg1 = x2b[(size_t)gv * 64 + c1];
    const float h0 = (float)k2buf[row + c0] - q2v + xg0;
    const float h1 = (float)k2buf[row + c1] - q2v + xg1;
    // v2 = t2 @ Wpv + bpv   (fused matvec)
    float v20 = sBPV[c0], v21 = sBPV[c1];
    for (int j = 0; j < 64; ++j) {
      const float tj = sT[wv][j];
      v20 += tj * sWPV[j * 64 + c0];
      v21 += tj * sWPV[j * 64 + c1];
    }
    sVP[wv][t][c0] = v20 + xg0;
    sVP[wv][t][c1] = v21 + xg1;

    const float s  = wave_sum32(h0 + h1);
    const float sq = wave_sum32(h0 * h0 + h1 * h1);
    const float mean = s * (1.f / 64.f);
    const float var  = sq * (1.f / 64.f) - mean * mean;
    const float inv  = rsqrtf(var + 1e-5f);
    sH[wv][c0] = fmaxf((h0 - mean) * inv * sG1[c0] + sBn1[c0], 0.f);
    sH[wv][c1] = fmaxf((h1 - mean) * inv * sG1[c1] + sBn1[c1], 0.f);
    __syncthreads();

    float lg = 0.f;
    if (ln < 8) {
      for (int c = 0; c < 64; ++c) lg += sH[wv][c] * sW1[c * 8 + ln];
      lg += sB1[ln];
    }
    const float s2  = oct_sum((ln < 8) ? lg : 0.f);
    const float q2s = oct_sum((ln < 8) ? lg * lg : 0.f);
    const float m2   = s2 * 0.125f;
    const float v2v  = q2s * 0.125f - m2 * m2;
    const float inv2 = rsqrtf(v2v + 1e-5f);
    float hn2 = 0.f;
    if (ln < 8) hn2 = fmaxf((lg - m2) * inv2 * sG2[ln] + sBn2[ln], 0.f);
    float outj = 0.f;
#pragma unroll
    for (int i = 0; i < 8; ++i) {
      const float hv = __shfl(hn2, i, 8);
      if (ln < 8) outj += hv * sW2s[i * 8 + ln];
    }
    if (ln < 8) sLog[wv][t][ln] = outj + sB2w[ln];
    __syncthreads();
  }

  if (ln < 8) {
    float mx = -1e30f;
    for (int t = 0; t < NSAMP; ++t) mx = fmaxf(mx, sLog[wv][t][ln]);
    float ssum = 0.f;
    for (int t = 0; t < NSAMP; ++t) { const float e = __expf(sLog[wv][t][ln] - mx); sLog[wv][t][ln] = e; ssum += e; }
    const float invs = 1.f / ssum;
    for (int t = 0; t < NSAMP; ++t) sLog[wv][t][ln] *= invs;
  }
  __syncthreads();

#pragma unroll
  for (int cc = 0; cc < 2; ++cc) {
    const int c = ln + 32 * cc;
    float acc = 0.f;
    for (int t = 0; t < NSAMP; ++t) acc += sVP[wv][t][c] * sLog[wv][t][c & 7];
    postbuf[(size_t)n * 64 + c] = acc;
    atomicAdd(&sBNs[c], acc);
    atomicAdd(&sBNq[c], acc * acc);
  }
  __syncthreads();
  if (tid < 64) { atomicAdd(&bnsum[tid], sBNs[tid]); atomicAdd(&bnsq[tid], sBNq[tid]); }
}

// ---------------------------------------------------------------------------
// Kbn: finalize batch-norm scale/shift; 1 block, 64 threads
// ---------------------------------------------------------------------------
__global__ void kbn_final(const float* __restrict__ fsum, const float* __restrict__ fsq,
                          const float* __restrict__ psum, const float* __restrict__ psq,
                          const float* __restrict__ gf, const float* __restrict__ bf,
                          const float* __restrict__ gp, const float* __restrict__ bp,
                          float* __restrict__ fscale, float* __restrict__ fshift,
                          float* __restrict__ pscale, float* __restrict__ pshift)
{
  const int c = threadIdx.x;
  const float invN = 1.f / (float)NPTS;
  const float mf = fsum[c] * invN;
  const float vf = fsq[c] * invN - mf * mf;
  const float sf = gf[c] * rsqrtf(vf + 1e-5f);
  fscale[c] = sf; fshift[c] = bf[c] - mf * sf;
  const float mp = psum[c] * invN;
  const float vp = psq[c] * invN - mp * mp;
  const float sp = gp[c] * rsqrtf(vp + 1e-5f);
  pscale[c] = sp; pshift[c] = bp[c] - mp * sp;
}

// ---------------------------------------------------------------------------
// K6: out = [bnrelu(feat) | bnrelu(post)] @ Wfp + bfp  (WMMA, K=128)
// grid NPTS/32, block 256
// ---------------------------------------------------------------------------
__global__ void k6_final(const float* __restrict__ feat, const float* __restrict__ post,
                         const float* __restrict__ fscale, const float* __restrict__ fshift,
                         const float* __restrict__ pscale, const float* __restrict__ pshift,
                         const float* __restrict__ Wfp, const float* __restrict__ bfp,
                         float* __restrict__ out)
{
  __shared__ __attribute__((aligned(16))) _Float16 As[32 * 136];
  __shared__ __attribute__((aligned(32))) _Float16 BsF[128 * 64];
  __shared__ float sBias[64];
  const int tid = threadIdx.x;
  const size_t r0 = (size_t)blockIdx.x * 32;

  for (int i = tid; i < 32 * 64; i += 256) {
    const int row = i >> 6, c = i & 63;
    const float f = fmaxf(feat[(r0 + row) * 64 + c] * fscale[c] + fshift[c], 0.f);
    const float p = fmaxf(post[(r0 + row) * 64 + c] * pscale[c] + pshift[c], 0.f);
    As[row * 136 + c]      = (_Float16)f;
    As[row * 136 + 64 + c] = (_Float16)p;
  }
  stage_b_frag<128>(Wfp, BsF, tid);
  if (tid < 64) sBias[tid] = bfp[tid];
  __syncthreads();

  const int wv = tid >> 5, lane = tid & 31;
  const int mBase = (wv & 1) * 16, ctile = wv >> 1;
  v8f acc = wmma_tile_frag<128>(As, 136, BsF, mBase, ctile, lane);
  const int hi = lane >> 4, nn = lane & 15;
#pragma unroll
  for (int r = 0; r < 8; ++r) {
    const int row = mBase + r + 8 * hi;
    out[(r0 + row) * 64 + ctile * 16 + nn] = acc[r] + sBias[ctile * 16 + nn];
  }
}

// ---------------------------------------------------------------------------
extern "C" void kernel_launch(void* const* d_in, const int* in_sizes, int n_in,
                              void* d_out, int out_size, void* d_ws, size_t ws_size,
                              hipStream_t stream)
{
  (void)in_sizes; (void)n_in; (void)out_size; (void)ws_size;
  const float* p    = (const float*)d_in[0];
  const float* x    = (const float*)d_in[1];
  const float* Wq   = (const float*)d_in[2];  const float* bq   = (const float*)d_in[3];
  const float* Wk   = (const float*)d_in[4];  const float* bk   = (const float*)d_in[5];
  const float* Wv   = (const float*)d_in[6];  const float* bv   = (const float*)d_in[7];
  const float* Wpq  = (const float*)d_in[8];  const float* bpq  = (const float*)d_in[9];
  const float* Wpk  = (const float*)d_in[10]; const float* bpk  = (const float*)d_in[11];
  const float* Wpv  = (const float*)d_in[12]; const float* bpv  = (const float*)d_in[13];
  const float* pW1  = (const float*)d_in[14]; const float* pb1  = (const float*)d_in[15];
  const float* plng = (const float*)d_in[16]; const float* plnb = (const float*)d_in[17];
  const float* pW2  = (const float*)d_in[18]; const float* pb2  = (const float*)d_in[19];
  const float* Wx   = (const float*)d_in[20]; const float* bx   = (const float*)d_in[21];
  const float* p2ln1g = (const float*)d_in[22]; const float* p2ln1b = (const float*)d_in[23];
  const float* p2W  = (const float*)d_in[24]; const float* p2b  = (const float*)d_in[25];
  const float* p2ln2g = (const float*)d_in[26]; const float* p2ln2b = (const float*)d_in[27];
  const float* fwln1g = (const float*)d_in[28]; const float* fwln1b = (const float*)d_in[29];
  const float* fwW1 = (const float*)d_in[30]; const float* fwb1 = (const float*)d_in[31];
  const float* fwln2g = (const float*)d_in[32]; const float* fwln2b = (const float*)d_in[33];
  const float* fwW2 = (const float*)d_in[34]; const float* fwb2 = (const float*)d_in[35];
  const float* pwln1g = (const float*)d_in[36]; const float* pwln1b = (const float*)d_in[37];
  const float* pwW1 = (const float*)d_in[38]; const float* pwb1 = (const float*)d_in[39];
  const float* pwln2g = (const float*)d_in[40]; const float* pwln2b = (const float*)d_in[41];
  const float* pwW2 = (const float*)d_in[42]; const float* pwb2 = (const float*)d_in[43];
  const float* brfg = (const float*)d_in[44]; const float* brfb = (const float*)d_in[45];
  const float* brpg = (const float*)d_in[46]; const float* brpb = (const float*)d_in[47];
  const float* Wfp  = (const float*)d_in[48]; const float* bfp  = (const float*)d_in[49];
  const int*   idx  = (const int*)d_in[50];
  float* out = (float*)d_out;

  // ---- workspace carve-up ----
  char* w = (char*)d_ws;
  auto take = [&](size_t bytes) { char* r = w; w += (bytes + 255) & ~(size_t)255; return r; };
  float*    qb    = (float*)take((size_t)NPTS * 64 * 4);
  float*    kb    = (float*)take((size_t)NPTS * 64 * 4);
  float*    vb    = (float*)take((size_t)NPTS * 64 * 4);
  float*    x2b   = (float*)take((size_t)NPTS * 64 * 4);
  _Float16* B1    = (_Float16*)take((size_t)MROWS * 64 * 2);  // pr -> t1 -> k2
  _Float16* B2    = (_Float16*)take((size_t)MROWS * 64 * 2);  // pr2 (t2)
  float*    q2b   = (float*)take((size_t)MROWS * 4);
  float*    featb = (float*)take((size_t)NPTS * 64 * 4);
  float*    postb = (float*)take((size_t)NPTS * 64 * 4);
  float*    bnAcc = (float*)take(256 * 4);     // fsum|fsq|psum|psq
  float*    bnSS  = (float*)take(256 * 4);     // fscale|fshift|pscale|pshift
  float*    wpqbar = (float*)take(64 * 4);
  float*    q2bias = (float*)take(4);

  float* fsum = bnAcc;       float* fsq = bnAcc + 64;
  float* psum = bnAcc + 128; float* psq = bnAcc + 192;
  float* fscale = bnSS;       float* fshift = bnSS + 64;
  float* pscale = bnSS + 128; float* pshift = bnSS + 192;

  // ---- pipeline ----
  hipLaunchKernelGGL(k0_prep, dim3(1), dim3(128), 0, stream, Wpq, bpq, wpqbar, q2bias, bnAcc);
  hipLaunchKernelGGL(k1_qkvx, dim3(NPTS / 32, 4), dim3(256), 0, stream,
                     x, Wq, bq, qb, Wk, bk, kb, Wv, bv, vb, Wx, bx, x2b);
  hipLaunchKernelGGL(k2_pr, dim3(MROWS / 256), dim3(256), 0, stream,
                     p, idx, pW1, pb1, plng, plnb, pW2, pb2, B1);
  hipLaunchKernelGGL(k3_feat, dim3(NPTS / 8), dim3(256), 0, stream,
                     qb, kb, vb, B1, idx,
                     fwln1g, fwln1b, fwW1, fwb1, fwln2g, fwln2b, fwW2, fwb2,
                     featb, fsum, fsq);
  hipLaunchKernelGGL(k4a_lnrelu, dim3(MROWS / 8), dim3(256), 0, stream, B1, p2ln1g, p2ln1b);
  hipLaunchKernelGGL(k4b_gemm_ln, dim3(MROWS / 32), dim3(256), 0, stream,
                     B1, p2W, p2b, p2ln2g, p2ln2b, B2);
  hipLaunchKernelGGL(k4d_gemm_q2, dim3(MROWS / 32), dim3(256), 0, stream,
                     B2, Wpk, bpk, B1, wpqbar, q2bias, q2b);
  hipLaunchKernelGGL(k5_post, dim3(NPTS / 8), dim3(256), 0, stream,
                     x2b, B1, B2, q2b, idx, Wpv, bpv,
                     pwln1g, pwln1b, pwW1, pwb1, pwln2g, pwln2b, pwW2, pwb2,
                     postb, psum, psq);
  hipLaunchKernelGGL(kbn_final, dim3(1), dim3(64), 0, stream,
                     fsum, fsq, psum, psq, brfg, brfb, brpg, brpb,
                     fscale, fshift, pscale, pshift);
  hipLaunchKernelGGL(k6_final, dim3(NPTS / 32), dim3(256), 0, stream,
                     featb, postb, fscale, fshift, pscale, pshift, Wfp, bfp, out);
}
